// Encoder_2113123909824
// MI455X (gfx1250) — matrix operand order repaired and checked by reading, not compiled, not run on hardware
//
#include <hip/hip_runtime.h>
#include <hip/hip_bf16.h>

typedef __bf16 bf16_t;
typedef __attribute__((ext_vector_type(16))) __bf16 v16bf;
typedef __attribute__((ext_vector_type(8)))  __bf16 v8bf;
typedef __attribute__((ext_vector_type(8)))  float  v8f;

// ---------------------------------------------------------------------------
// WMMA bf16 GEMM: C[M,N] = A[M,K] @ B[K,N] (+bias), B given transposed [N,K].
// Block = 8 waves = 128 rows x 64 cols macro-tile. The 64xK B slab is staged
// into LDS once per block with gfx1250 async global->LDS copies (ASYNCcnt),
// then each wave runs a ds_load + global_load + v_wmma pipeline.
// M%16==0, N%64==0, K%32==0, K<=256 guaranteed by the call sites.
// ---------------------------------------------------------------------------
__global__ __launch_bounds__(256) void gemm_bf16_wmma(
    const bf16_t* __restrict__ A,   // [M,K] row-major bf16
    const bf16_t* __restrict__ BT,  // [N,K] row-major bf16 (B transposed)
    const float*  __restrict__ bias,// [N] or nullptr
    float*        __restrict__ C,   // [M,N] fp32
    int M, int N, int K)
{
    __shared__ bf16_t Bs[64 * 264];   // [col][K+8] padded; 33,792 B max (K=256)

    const int tid  = threadIdx.x;
    const int lane = tid & 31;
    const int wave = tid >> 5;
    const int half = lane >> 4;       // 0: lanes 0-15, 1: lanes 16-31
    const int l15  = lane & 15;

    const int tilesN  = N >> 6;
    const int bm      = blockIdx.x / tilesN;  // macro row-block (8 row tiles)
    const int tn      = blockIdx.x % tilesN;
    const int colbase = tn << 6;
    const int ldsK    = K + 8;                // padded column stride (elements)

    // ---- stage B slab [64 x K] into LDS via async global->LDS b128 copies ----
    {
        const unsigned ldsBase = (unsigned)(uintptr_t)&Bs[0]; // low 32 bits = LDS offset
        const int chunksPerCol = K >> 3;                      // 16B chunks per column
        const int chunks = 64 * chunksPerCol;
        for (int i = tid; i < chunks; i += 256) {
            int col = i / chunksPerCol;
            int ko  = (i - col * chunksPerCol) << 3;
            const bf16_t* gp = BT + (size_t)(colbase + col) * K + ko;
            unsigned lo = ldsBase + (unsigned)(col * ldsK + ko) * 2u;
            asm volatile("global_load_async_to_lds_b128 %0, %1, off"
                         :: "v"(lo), "v"(gp)
                         : "memory");
        }
        asm volatile("s_wait_asynccnt 0" ::: "memory");
        __syncthreads();
    }

    // ---- wave-uniform exit for the ragged last row-block (after the barrier) ----
    const int rowTile = bm * 8 + wave;
    if (rowTile >= (M >> 4)) return;

    const int arow = (rowTile << 4) + l15;
    const bf16_t* Arow = A + (size_t)arow * K;
    const bf16_t* Bs0 = &Bs[(l15 +  0) * ldsK];
    const bf16_t* Bs1 = &Bs[(l15 + 16) * ldsK];
    const bf16_t* Bs2 = &Bs[(l15 + 32) * ldsK];
    const bf16_t* Bs3 = &Bs[(l15 + 48) * ldsK];

    v8f acc0 = {}, acc1 = {}, acc2 = {}, acc3 = {};

    for (int kk = 0; kk < K; kk += 32) {
        __builtin_prefetch((const void*)(Arow + kk + 32), 0, 3);
        // A fragment (16x32 bf16 layout): contiguous K run [kk+half*16,+16) as two 8-chunks
        v8bf a0 = *(const v8bf*)(Arow + kk +     half * 16);
        v8bf a1 = *(const v8bf*)(Arow + kk + 8 + half * 16);
        v16bf a = __builtin_shufflevector(a0, a1, 0,1,2,3,4,5,6,7,8,9,10,11,12,13,14,15);
        // B fragments (32x16 bf16 layout): contiguous K run [kk+half*16,+16) per column
        v16bf b0 = *(const v16bf*)(Bs0 + kk + half * 16);
        v16bf b1 = *(const v16bf*)(Bs1 + kk + half * 16);
        v16bf b2 = *(const v16bf*)(Bs2 + kk + half * 16);
        v16bf b3 = *(const v16bf*)(Bs3 + kk + half * 16);
        acc0 = __builtin_amdgcn_wmma_f32_16x16x32_bf16(false, a, false, b0, (short)0, acc0, false, false);
        acc1 = __builtin_amdgcn_wmma_f32_16x16x32_bf16(false, a, false, b1, (short)0, acc1, false, false);
        acc2 = __builtin_amdgcn_wmma_f32_16x16x32_bf16(false, a, false, b2, (short)0, acc2, false, false);
        acc3 = __builtin_amdgcn_wmma_f32_16x16x32_bf16(false, a, false, b3, (short)0, acc3, false, false);
    }

    float bv0 = 0.f, bv1 = 0.f, bv2 = 0.f, bv3 = 0.f;
    if (bias) {
        bv0 = bias[colbase +  0 + l15];
        bv1 = bias[colbase + 16 + l15];
        bv2 = bias[colbase + 32 + l15];
        bv3 = bias[colbase + 48 + l15];
    }
    // C layout: VGPR r -> row rowTile*16 + half*8 + r, col = colbase + j*16 + l15
    const int crowbase = (rowTile << 4) + half * 8;
#pragma unroll
    for (int r = 0; r < 8; r++) {
        size_t o = (size_t)(crowbase + r) * N + colbase + l15;
        C[o +  0] = acc0[r] + bv0;
        C[o + 16] = acc1[r] + bv1;
        C[o + 32] = acc2[r] + bv2;
        C[o + 48] = acc3[r] + bv3;
    }
}

// ---------------------------------------------------------------------------
// Elementwise / graph kernels
// ---------------------------------------------------------------------------
__global__ void f32_to_bf16(const float* __restrict__ in, bf16_t* __restrict__ out, int n) {
    int t = blockIdx.x * blockDim.x + threadIdx.x;
    if (t < n) out[t] = (bf16_t)in[t];
}

__global__ void transpose_to_bf16(const float* __restrict__ W, bf16_t* __restrict__ WT, int K, int N) {
    // W[K,N] row-major -> WT[N,K] row-major bf16
    int idx = blockIdx.x * blockDim.x + threadIdx.x;
    if (idx >= K * N) return;
    int n = idx / K, k = idx % K;
    WT[idx] = (bf16_t)W[(size_t)k * N + n];
}

__global__ void fill_f32(float* __restrict__ p, float v, int n) {
    int t = blockIdx.x * blockDim.x + threadIdx.x;
    if (t < n) p[t] = v;
}

// agg[si[e]] += x[gi[e]] * ew[e] over 256 features (4 per thread)
__global__ void gcn_scatter(const float* __restrict__ x, const float* __restrict__ ew,
                            const int* __restrict__ gi, const int* __restrict__ si,
                            float* __restrict__ agg, int E) {
    int t = blockIdx.x * blockDim.x + threadIdx.x;
    if (t >= E * 64) return;
    int e = t >> 6;
    int c = (t & 63) << 2;
    float w = ew[e];
    const float* xs = x + (size_t)gi[e] * 256 + c;
    float* ad = agg + (size_t)si[e] * 256 + c;
    atomicAdd(ad + 0, xs[0] * w);
    atomicAdd(ad + 1, xs[1] * w);
    atomicAdd(ad + 2, xs[2] * w);
    atomicAdd(ad + 3, xs[3] * w);
}

__global__ void add_relu(float* __restrict__ H, const float* __restrict__ T, int n) {
    int t = blockIdx.x * blockDim.x + threadIdx.x;
    if (t < n) { float v = T[t]; H[t] += (v > 0.f ? v : 0.f); }
}

// hidden = feature + hidden * sigmoid(feature), in place on H
__global__ void identity_combine(const float* __restrict__ feat, float* __restrict__ H, int n) {
    int t = blockIdx.x * blockDim.x + threadIdx.x;
    if (t < n) {
        float f = feat[t];
        float s = 1.f / (1.f + __expf(-f));
        H[t] = f + H[t] * s;
    }
}

// el[n,h] = sum_d FS[n,h*128+d]*al[h,d]; er likewise
__global__ void head_dots(const float* __restrict__ FS, const float* __restrict__ al,
                          const float* __restrict__ ar, float* __restrict__ EL,
                          float* __restrict__ ER, int n) {
    int t = blockIdx.x * blockDim.x + threadIdx.x;
    if (t >= n * 4) return;
    int node = t >> 2, h = t & 3;
    const float* f = FS + (size_t)node * 512 + h * 128;
    const float* a = al + h * 128;
    const float* b = ar + h * 128;
    float sl = 0.f, sr = 0.f;
    for (int d = 0; d < 128; d++) { float v = f[d]; sl += v * a[d]; sr += v * b[d]; }
    EL[t] = sl; ER[t] = sr;
}

__device__ inline void atomicMaxF(float* addr, float val) {
    int* ai = (int*)addr;
    int cur = *ai;
    while (__int_as_float(cur) < val) {
        int prev = atomicCAS(ai, cur, __float_as_int(val));
        if (prev == cur) break;
        cur = prev;
    }
}

// z = leaky_relu(EL[S]+ER[D]); ELOG=z; MX[D]=max
__global__ void edge_logit_max(const float* __restrict__ EL, const float* __restrict__ ER,
                               const int* __restrict__ S, const int* __restrict__ D,
                               float* __restrict__ ELOG, float* __restrict__ MX, int E) {
    int t = blockIdx.x * blockDim.x + threadIdx.x;
    if (t >= E * 4) return;
    int e = t >> 2, h = t & 3;
    float z = EL[S[e] * 4 + h] + ER[D[e] * 4 + h];
    z = (z > 0.f) ? z : 0.2f * z;
    ELOG[t] = z;
    atomicMaxF(&MX[D[e] * 4 + h], z);
}

// ex = exp(z - MX[D]); ELOG=ex; SM[D]+=ex
__global__ void edge_exp_sum(float* __restrict__ ELOG, const float* __restrict__ MX,
                             float* __restrict__ SM, const int* __restrict__ D, int E) {
    int t = blockIdx.x * blockDim.x + threadIdx.x;
    if (t >= E * 4) return;
    int e = t >> 2, h = t & 3;
    float ex = __expf(ELOG[t] - MX[D[e] * 4 + h]);
    ELOG[t] = ex;
    atomicAdd(&SM[D[e] * 4 + h], ex);
}

// OUT[D] += FS[S] * (ex / SM[D])  over [H=4, D=128], 4 floats per thread
__global__ void edge_message(const float* __restrict__ FS, const float* __restrict__ ELOG,
                             const float* __restrict__ SM, const int* __restrict__ S,
                             const int* __restrict__ D, float* __restrict__ OUT, int E) {
    int t = blockIdx.x * blockDim.x + threadIdx.x;
    if (t >= E * 128) return;
    int e = t >> 7;
    int c = (t & 127) << 2;   // 0..508
    int h = c >> 7;           // head index
    int s = S[e], d = D[e];
    float a = ELOG[e * 4 + h] / SM[d * 4 + h];
    const float* fs = FS + (size_t)s * 512 + c;
    float* op = OUT + (size_t)d * 512 + c;
    atomicAdd(op + 0, fs[0] * a);
    atomicAdd(op + 1, fs[1] * a);
    atomicAdd(op + 2, fs[2] * a);
    atomicAdd(op + 3, fs[3] * a);
}

// m = mean over 4 heads; mode0: XB=bf16(m); mode1: H2+=m, XB=bf16(H2)
__global__ void block_combine(const float* __restrict__ OUT, float* __restrict__ H2,
                              bf16_t* __restrict__ XB, int n, int addResidual) {
    int t = blockIdx.x * blockDim.x + threadIdx.x;
    if (t >= n * 128) return;
    int node = t >> 7, d = t & 127;
    const float* o = OUT + (size_t)node * 512 + d;
    float m = 0.25f * (o[0] + o[128] + o[256] + o[384]);
    if (addResidual) { float h = H2[t] + m; H2[t] = h; XB[t] = (bf16_t)h; }
    else             { XB[t] = (bf16_t)m; }
}

// Per-feature mean / inv-std over 20000 nodes (one block per feature)
__global__ __launch_bounds__(256) void bn_stats(const float* __restrict__ H2,
                                                float* __restrict__ stats, int n) {
    __shared__ float ss[256];
    __shared__ float sq[256];
    int f = blockIdx.x;
    float s = 0.f, q = 0.f;
    for (int i = threadIdx.x; i < n; i += 256) {
        float v = H2[(size_t)i * 128 + f];
        s += v; q += v * v;
    }
    ss[threadIdx.x] = s; sq[threadIdx.x] = q;
    __syncthreads();
    for (int st = 128; st > 0; st >>= 1) {
        if (threadIdx.x < st) { ss[threadIdx.x] += ss[threadIdx.x + st]; sq[threadIdx.x] += sq[threadIdx.x + st]; }
        __syncthreads();
    }
    if (threadIdx.x == 0) {
        float mu  = ss[0] / n;
        float var = sq[0] / n - mu * mu;
        stats[f]       = mu;
        stats[128 + f] = rsqrtf(var + 1e-5f);
    }
}

__global__ void bn_apply(const float* __restrict__ H2, const float* __restrict__ stats,
                         const float* __restrict__ gamma, const float* __restrict__ beta,
                         float* __restrict__ out, int total) {
    int t = blockIdx.x * blockDim.x + threadIdx.x;
    if (t >= total) return;
    int f = t & 127;
    out[t] = gamma[f] * (H2[t] - stats[f]) * stats[128 + f] + beta[f];
}

// ---------------------------------------------------------------------------
static void launch_gemm(const bf16_t* A, const bf16_t* BT, const float* bias, float* C,
                        int M, int Ncols, int K, hipStream_t stream) {
    int rowBlocks = ((M / 16) + 7) / 8;          // 8 row-tiles (128 rows) per block
    int blocks = rowBlocks * (Ncols / 64);
    gemm_bf16_wmma<<<blocks, 256, 0, stream>>>(A, BT, bias, C, M, Ncols, K);
}

extern "C" void kernel_launch(void* const* d_in, const int* in_sizes, int n_in,
                              void* d_out, int out_size, void* d_ws, size_t ws_size,
                              hipStream_t stream) {
    (void)in_sizes; (void)n_in; (void)out_size; (void)ws_size;
    const int N = 20000, E = 160000;

    const float* feature  = (const float*)d_in[0];
    const float* ew_f     = (const float*)d_in[1];
    const float* ew_b     = (const float*)d_in[2];
    const float* W_id     = (const float*)d_in[3];
    const float* b_id     = (const float*)d_in[4];
    const float* W_gcnf   = (const float*)d_in[5];
    const float* b_gcnf   = (const float*)d_in[6];
    const float* W_gcnb   = (const float*)d_in[7];
    const float* b_gcnb   = (const float*)d_in[8];
    const float* W1       = (const float*)d_in[9];
    const float* b1       = (const float*)d_in[10];
    const float* gat_W    = (const float*)d_in[11];
    const float* gat_al   = (const float*)d_in[12];
    const float* gat_ar   = (const float*)d_in[13];
    const float* bn_gamma = (const float*)d_in[14];
    const float* bn_beta  = (const float*)d_in[15];
    const int*   src      = (const int*)d_in[16];
    const int*   dst      = (const int*)d_in[17];
    float* out = (float*)d_out;

    // ---- workspace layout ----
    char* ws = (char*)d_ws;
    size_t off = 0;
    auto alloc = [&](size_t bytes) -> void* {
        void* p = ws + off;
        off += (bytes + 255) & ~(size_t)255;
        return p;
    };
    bf16_t* BF_X  = (bf16_t*)alloc((size_t)N * 256 * 2);   // bf16 activation scratch
    bf16_t* BF_W  = (bf16_t*)alloc((size_t)753664 * 2);    // all transposed bf16 weights
    float*  AGG   = (float*)alloc((size_t)N * 256 * 4);
    float*  H     = (float*)alloc((size_t)N * 256 * 4);
    float*  T     = (float*)alloc((size_t)N * 512 * 4);    // GEMM output / FS
    float*  OUT   = (float*)alloc((size_t)N * 512 * 4);    // per-block head accumulator
    float*  H2    = (float*)alloc((size_t)N * 128 * 4);    // residual stream
    float*  EL    = (float*)alloc((size_t)N * 4 * 4);
    float*  ER    = (float*)alloc((size_t)N * 4 * 4);
    float*  MX    = (float*)alloc((size_t)N * 4 * 4);
    float*  SM    = (float*)alloc((size_t)N * 4 * 4);
    float*  ELOG  = (float*)alloc((size_t)E * 4 * 4);
    float*  STATS = (float*)alloc(256 * 4);

    bf16_t* WT_id  = BF_W;
    bf16_t* WT_gf  = BF_W + 65536;
    bf16_t* WT_gb  = BF_W + 131072;
    bf16_t* WT_1   = BF_W + 196608;   // [128,256]
    bf16_t* WT_gat = BF_W + 229376;   // 8 x [512,128]

    const int TB = 256;
    auto g1 = [](int total) { return dim3((unsigned)((total + 255) / 256)); };

    // ---- weight transposes (fp32 -> bf16 [N,K]) ----
    transpose_to_bf16<<<g1(256 * 256), TB, 0, stream>>>(W_id,   WT_id, 256, 256);
    transpose_to_bf16<<<g1(256 * 256), TB, 0, stream>>>(W_gcnf, WT_gf, 256, 256);
    transpose_to_bf16<<<g1(256 * 256), TB, 0, stream>>>(W_gcnb, WT_gb, 256, 256);
    transpose_to_bf16<<<g1(256 * 128), TB, 0, stream>>>(W1,     WT_1,  256, 128);
    for (int i = 0; i < 8; i++)
        transpose_to_bf16<<<g1(128 * 512), TB, 0, stream>>>(
            gat_W + (size_t)i * 128 * 512, WT_gat + (size_t)i * 65536, 128, 512);

    // ---- IdentityLayer: lin + fwd WGCN + bwd WGCN ----
    f32_to_bf16<<<g1(N * 256), TB, 0, stream>>>(feature, BF_X, N * 256);
    launch_gemm(BF_X, WT_id, b_id, H, N, 256, 256, stream);        // H = feat@W_id + b

    fill_f32<<<g1(N * 256), TB, 0, stream>>>(AGG, 0.f, N * 256);
    gcn_scatter<<<g1(E * 64), TB, 0, stream>>>(feature, ew_f, src, dst, AGG, E);
    f32_to_bf16<<<g1(N * 256), TB, 0, stream>>>(AGG, BF_X, N * 256);
    launch_gemm(BF_X, WT_gf, b_gcnf, T, N, 256, 256, stream);
    add_relu<<<g1(N * 256), TB, 0, stream>>>(H, T, N * 256);       // H += relu(.)

    fill_f32<<<g1(N * 256), TB, 0, stream>>>(AGG, 0.f, N * 256);
    gcn_scatter<<<g1(E * 64), TB, 0, stream>>>(feature, ew_b, dst, src, AGG, E);
    f32_to_bf16<<<g1(N * 256), TB, 0, stream>>>(AGG, BF_X, N * 256);
    launch_gemm(BF_X, WT_gb, b_gcnb, T, N, 256, 256, stream);
    add_relu<<<g1(N * 256), TB, 0, stream>>>(H, T, N * 256);

    identity_combine<<<g1(N * 256), TB, 0, stream>>>(feature, H, N * 256);
    f32_to_bf16<<<g1(N * 256), TB, 0, stream>>>(H, BF_X, N * 256);
    launch_gemm(BF_X, WT_1, b1, H2, N, 128, 256, stream);          // H2 = hidden@W1 + b1

    // ---- GAT layers ----
    f32_to_bf16<<<g1(N * 128), TB, 0, stream>>>(H2, BF_X, N * 128);
    for (int l = 0; l < 2; l++) {
        for (int blk = 0; blk < 2; blk++) {
            fill_f32<<<g1(N * 512), TB, 0, stream>>>(OUT, 0.f, N * 512);
            for (int dir = 0; dir < 2; dir++) {
                int wi = (l * 2 + blk) * 2 + dir;
                launch_gemm(BF_X, WT_gat + (size_t)wi * 65536, nullptr, T, N, 512, 128, stream);
                head_dots<<<g1(N * 4), TB, 0, stream>>>(T, gat_al + (size_t)wi * 512,
                                                        gat_ar + (size_t)wi * 512, EL, ER, N);
                fill_f32<<<g1(N * 4), TB, 0, stream>>>(MX, -1e30f, N * 4);
                fill_f32<<<g1(N * 4), TB, 0, stream>>>(SM, 0.f, N * 4);
                const int* S = (dir == 0) ? src : dst;
                const int* D = (dir == 0) ? dst : src;
                edge_logit_max<<<g1(E * 4), TB, 0, stream>>>(EL, ER, S, D, ELOG, MX, E);
                edge_exp_sum<<<g1(E * 4), TB, 0, stream>>>(ELOG, MX, SM, D, E);
                edge_message<<<g1(E * 128), TB, 0, stream>>>(T, ELOG, SM, S, D, OUT, E);
            }
            block_combine<<<g1(N * 128), TB, 0, stream>>>(OUT, H2, BF_X, N, blk == 1 ? 1 : 0);
        }
    }

    // ---- BatchNorm (training-mode batch stats) ----
    bn_stats<<<128, 256, 0, stream>>>(H2, STATS, N);
    bn_apply<<<g1(N * 128), TB, 0, stream>>>(H2, STATS, bn_gamma, bn_beta, out, N * 128);
}
